// ReactionCenterPredictor_80290118631443
// MI455X (gfx1250) — matrix operand order repaired and checked by reading, not compiled, not run on hardware
//
#include <hip/hip_runtime.h>
#include <hip/hip_bf16.h>

// ---------------------------------------------------------------------------
// ReactionCenterPredictor GNN on gfx1250 (MI455X), bf16 WMMA pipeline.
//   - edge message MLP [E,128]@[128,64] and update MLP [N,128]@[128,64]
//     run on v_wmma_f32_16x16x32_bf16 (f32 accumulate)
//   - weights staged to LDS (transposed bf16), A tiles gathered to LDS
//   - segment-sum via global_atomic_add_f32 (L2-resident, h fits in 192MB L2)
// ---------------------------------------------------------------------------

typedef __attribute__((ext_vector_type(16))) __bf16 v16bf;
typedef __attribute__((ext_vector_type(8)))  float  v8f;

union FragU { v16bf v; uint4 q[2]; };

__device__ __forceinline__ unsigned short f32_to_bf16(float f) {
    unsigned int u = __float_as_uint(f);
    unsigned int rnd = 0x7FFFu + ((u >> 16) & 1u);
    u += rnd;
    return (unsigned short)(u >> 16);
}

// Build a 16x bf16 fragment from two contiguous 8-element (16B) runs.
__device__ __forceinline__ v16bf make_frag(const unsigned short* p0,
                                           const unsigned short* p1) {
    FragU f;
    f.q[0] = *(const uint4*)p0;
    f.q[1] = *(const uint4*)p1;
    return f.v;
}

__device__ __forceinline__ v8f wmma_bf16(v16bf a, v16bf b, v8f c) {
    return __builtin_amdgcn_wmma_f32_16x16x32_bf16(
        false, a, false, b, (short)0, c, false, false);
}

// ---------------------------------------------------------------------------
// Weight prep: transpose + bf16 convert msg_W / upd_W.
// wT layout: [(l*2+mat)][n(64)][k(128)]  (mat 0 = msg, 1 = upd)
// ---------------------------------------------------------------------------
__global__ void k_wprep(const float* __restrict__ msgW,
                        const float* __restrict__ updW,
                        unsigned short* __restrict__ wT, int total) {
    int gid = blockIdx.x * blockDim.x + threadIdx.x;
    if (gid >= total) return;
    int k   = gid & 127;
    int n   = (gid >> 7) & 63;
    int mat = (gid >> 13) & 1;
    int l   = gid >> 14;
    const float* src = mat ? updW : msgW;
    wT[gid] = f32_to_bf16(src[((long long)l * 128 + k) * 64 + n]);
}

// ---------------------------------------------------------------------------
// Input GEMM: h = x @ W_in + b_in   ([N,39]@[39,64])
// ---------------------------------------------------------------------------
__global__ __launch_bounds__(256) void k_in(const float* __restrict__ x,
                                            const float* __restrict__ W,
                                            const float* __restrict__ b,
                                            float* __restrict__ h,
                                            unsigned short* __restrict__ hb,
                                            int Nn) {
    __shared__ float Ws[39 * 64];
    for (int i = threadIdx.x; i < 39 * 64; i += 256) Ws[i] = W[i];
    __syncthreads();
    long long gid = (long long)blockIdx.x * 256 + threadIdx.x;
    long long node = gid >> 6;
    int f = (int)(gid & 63);
    if (node >= Nn) return;
    const float* xr = x + node * 39;
    float acc = b[f];
#pragma unroll
    for (int k = 0; k < 39; ++k) acc = fmaf(xr[k], Ws[k * 64 + f], acc);
    h[node * 64 + f]  = acc;
    hb[node * 64 + f] = f32_to_bf16(acc);
}

// ---------------------------------------------------------------------------
// In-degree count over col
// ---------------------------------------------------------------------------
__global__ void k_degree(const long long* __restrict__ col,
                         float* __restrict__ cnt, long long E) {
    long long e = (long long)blockIdx.x * blockDim.x + threadIdx.x;
    if (e < E) atomicAdd(&cnt[col[e]], 1.0f);
}

// ---------------------------------------------------------------------------
// Edge message MLP + scatter-add:
//   msg = concat(h[row], h[col]) @ msg_W + msg_b ; aggr[col] += msg
// Block = 256 thr (8 waves). Weights staged once per block (16 KB LDS).
// Each block processes EDGE_TILES x 128 edges; each wave owns 16 edges/tile.
// ---------------------------------------------------------------------------
#define EDGE_TILES 4

__global__ __launch_bounds__(256) void k_edge(
    const unsigned short* __restrict__ hb, const long long* __restrict__ row,
    const long long* __restrict__ col, const unsigned short* __restrict__ wT,
    const float* __restrict__ bias, float* __restrict__ aggr, long long E) {
    __shared__ unsigned short a_lds[128 * 128];  // 32 KB: [edge][K]
    __shared__ unsigned short w_lds[64 * 128];   // 16 KB: W^T bf16 [n][k]

    const int tid = threadIdx.x;
    const long long blockStart = (long long)blockIdx.x * (128 * EDGE_TILES);

    // ---- Stage weights once per block ----
    {
        uint4* wd = (uint4*)w_lds;
        const uint4* wsrc = (const uint4*)wT;
#pragma unroll
        for (int i = 0; i < 4; ++i) wd[tid + 256 * i] = wsrc[tid + 256 * i];
    }

    const int wave = tid >> 5, lane = tid & 31;
    const int m = lane & 15, kh = lane >> 4;
    const int n0 = lane & 15;
    const unsigned short* arow = &a_lds[(wave * 16 + m) * 128];

    // Biases are loop-invariant
    float bvals[4];
#pragma unroll
    for (int nt = 0; nt < 4; ++nt) bvals[nt] = bias[nt * 16 + n0];

    for (int t = 0; t < EDGE_TILES; ++t) {
        const long long blockBase = blockStart + (long long)t * 128;
        if (blockBase >= E) break;  // uniform across block
        const bool fullTile = (blockBase + 128 <= E);

        // ---- Stage A: gather bf16 endpoint features (64 each) per edge ----
        {
            int e_local = tid >> 1;
            int half = tid & 1;  // 0: h[row], 1: h[col]
            long long e = blockBase + e_local;
            uint4* dst = (uint4*)&a_lds[e_local * 128 + half * 64];
            if (fullTile || e < E) {
                long long node = (half ? col : row)[e];
                const uint4* src = (const uint4*)(hb + node * 64);
#pragma unroll
                for (int i = 0; i < 8; ++i) dst[i] = src[i];
            } else {
                uint4 z = {0u, 0u, 0u, 0u};
#pragma unroll
                for (int i = 0; i < 8; ++i) dst[i] = z;
            }
            // Prefetch next tile's edge indices (global_prefetch_b8)
            long long nb = blockBase + 128 + e_local;
            if (t + 1 < EDGE_TILES && nb < E)
                __builtin_prefetch(half ? &col[nb] : &row[nb], 0, 0);
        }
        __syncthreads();

        // ---- A fragments: per 32-wide K step, lane holds contiguous K run
        //      [kh*16..+15] (matches B fragment order) ----
        v16bf a[4];
#pragma unroll
        for (int ks = 0; ks < 4; ++ks)
            a[ks] = make_frag(arow + 32 * ks + kh * 16,
                              arow + 32 * ks + kh * 16 + 8);

        // ---- 16 WMMAs: 4 N-tiles x 4 K-steps, bias folded into C ----
        v8f c4[4];
#pragma unroll
        for (int nt = 0; nt < 4; ++nt) {
#pragma unroll
            for (int i = 0; i < 8; ++i) c4[nt][i] = bvals[nt];
        }
#pragma unroll
        for (int nt = 0; nt < 4; ++nt) {
            const unsigned short* wb = w_lds + (nt * 16 + n0) * 128 + kh * 16;
#pragma unroll
            for (int ks = 0; ks < 4; ++ks) {
                v16bf b = make_frag(wb + 32 * ks, wb + 32 * ks + 8);
                c4[nt] = wmma_bf16(a[ks], b, c4[nt]);
            }
        }

        // ---- Scatter-add into aggr[col]  (C layout: VGPR r <-> M = r/r+8,
        //      feature = nt*16 + (lane&15)) ----
        const long long tileBase = blockBase + wave * 16;
        if (fullTile) {
#pragma unroll
            for (int r = 0; r < 8; ++r) {
                int M = (lane < 16) ? r : r + 8;
                long long dst = col[tileBase + M];
                float* ap = aggr + dst * 64 + n0;
#pragma unroll
                for (int nt = 0; nt < 4; ++nt)
                    atomicAdd(ap + nt * 16, c4[nt][r]);
            }
        } else {
#pragma unroll
            for (int r = 0; r < 8; ++r) {
                int M = (lane < 16) ? r : r + 8;
                long long e = tileBase + M;
                if (e < E) {
                    long long dst = col[e];
                    float* ap = aggr + dst * 64 + n0;
#pragma unroll
                    for (int nt = 0; nt < 4; ++nt)
                        atomicAdd(ap + nt * 16, c4[nt][r]);
                }
            }
        }
        __syncthreads();  // protect a_lds before next tile's staging
    }
}

// ---------------------------------------------------------------------------
// Update GEMM + residual + LayerNorm (in place on h / hb):
//   upd = concat(h, aggr/denom) @ upd_W + upd_b ;  h = LN(h + upd)*g + b
// Block tile = 128 nodes.
// ---------------------------------------------------------------------------
__global__ __launch_bounds__(256) void k_update(
    float* __restrict__ h, unsigned short* __restrict__ hb,
    const float* __restrict__ aggr, const float* __restrict__ cnt,
    const unsigned short* __restrict__ wT, const float* __restrict__ ubias,
    const float* __restrict__ lng, const float* __restrict__ lnb, int Nn) {
    __shared__ unsigned short a_lds[128 * 128];  // 32 KB
    __shared__ unsigned short w_lds[64 * 128];   // 16 KB
    __shared__ float u_lds[128 * 64];            // 32 KB

    const int tid = threadIdx.x;
    const long long base = (long long)blockIdx.x * 128;

    // ---- Stage weights ----
    {
        uint4* wd = (uint4*)w_lds;
        const uint4* wsrc = (const uint4*)wT;
#pragma unroll
        for (int i = 0; i < 4; ++i) wd[tid + 256 * i] = wsrc[tid + 256 * i];
    }

    // ---- Stage A = [ h_bf16 | bf16(aggr/denom) ] ----
    {
        int i_local = tid >> 1;
        int half = tid & 1;
        long long node = base + i_local;
        if (half == 0) {
            uint4* dst = (uint4*)&a_lds[i_local * 128];
            if (node < Nn) {
                const uint4* src = (const uint4*)(hb + node * 64);
#pragma unroll
                for (int i = 0; i < 8; ++i) dst[i] = src[i];
            } else {
                uint4 z = {0u, 0u, 0u, 0u};
#pragma unroll
                for (int i = 0; i < 8; ++i) dst[i] = z;
            }
        } else {
            unsigned short* dst = &a_lds[i_local * 128 + 64];
            if (node < Nn) {
                float rd = 1.0f / fmaxf(cnt[node], 1.0f);
                const float* ag = aggr + node * 64;
#pragma unroll
                for (int j = 0; j < 64; ++j) dst[j] = f32_to_bf16(ag[j] * rd);
            } else {
#pragma unroll
                for (int j = 0; j < 64; ++j) dst[j] = 0;
            }
        }
    }
    __syncthreads();

    const int wave = tid >> 5, lane = tid & 31;
    const int m = lane & 15, kh = lane >> 4;
    const int n0 = lane & 15;
    const unsigned short* arow = &a_lds[(wave * 16 + m) * 128];
    v16bf a[4];
#pragma unroll
    for (int ks = 0; ks < 4; ++ks)
        a[ks] = make_frag(arow + 32 * ks + kh * 16, arow + 32 * ks + kh * 16 + 8);

#pragma unroll
    for (int nt = 0; nt < 4; ++nt) {
        int n = nt * 16 + n0;
        float bi = ubias[n];
        v8f c;
#pragma unroll
        for (int i = 0; i < 8; ++i) c[i] = bi;
        const unsigned short* wb = w_lds + n * 128 + kh * 16;
#pragma unroll
        for (int ks = 0; ks < 4; ++ks) {
            v16bf b = make_frag(wb + 32 * ks, wb + 32 * ks + 8);
            c = wmma_bf16(a[ks], b, c);
        }
#pragma unroll
        for (int r = 0; r < 8; ++r) {
            int M = (lane < 16) ? r : r + 8;
            u_lds[(wave * 16 + M) * 64 + n] = c[r];
        }
    }
    __syncthreads();

    // ---- Residual + LayerNorm: one thread per node ----
    if (tid < 128) {
        long long node = base + tid;
        if (node < Nn) {
            float* u = &u_lds[tid * 64];
            const float* hg = h + node * 64;
            float sum = 0.f, sumsq = 0.f;
#pragma unroll
            for (int f = 0; f < 64; ++f) {
                float z = hg[f] + u[f];
                u[f] = z;
                sum += z;
                sumsq += z * z;
            }
            float mu = sum * 0.015625f;
            float var = sumsq * 0.015625f - mu * mu;
            float rstd = rsqrtf(var + 1e-5f);
#pragma unroll
            for (int f = 0; f < 64; ++f) {
                float res = (u[f] - mu) * rstd * lng[f] + lnb[f];
                h[node * 64 + f] = res;
                hb[node * 64 + f] = f32_to_bf16(res);
            }
        }
    }
}

// ---------------------------------------------------------------------------
// Output head: sigmoid(gelu(h@W1+b1)@W2 + b2)
// ---------------------------------------------------------------------------
__global__ __launch_bounds__(256) void k_out(const float* __restrict__ h,
                                             const float* __restrict__ W1,
                                             const float* __restrict__ b1,
                                             const float* __restrict__ W2,
                                             const float* __restrict__ b2,
                                             float* __restrict__ out, int Nn) {
    __shared__ float W1s[64 * 32];
    __shared__ float b1s[32];
    __shared__ float W2s[32];
    __shared__ float b2s;
    for (int i = threadIdx.x; i < 64 * 32; i += 256) W1s[i] = W1[i];
    if (threadIdx.x < 32) {
        b1s[threadIdx.x] = b1[threadIdx.x];
        W2s[threadIdx.x] = W2[threadIdx.x];
    }
    if (threadIdx.x == 0) b2s = b2[0];
    __syncthreads();

    long long node = (long long)blockIdx.x * 256 + threadIdx.x;
    if (node >= Nn) return;
    float hr[64];
#pragma unroll
    for (int k = 0; k < 64; ++k) hr[k] = h[node * 64 + k];
    float o = b2s;
#pragma unroll 4
    for (int j = 0; j < 32; ++j) {
        float acc = b1s[j];
#pragma unroll
        for (int k = 0; k < 64; ++k) acc = fmaf(hr[k], W1s[k * 32 + j], acc);
        float g = 0.5f * acc * (1.0f + erff(acc * 0.70710678118654752f));
        o = fmaf(g, W2s[j], o);
    }
    out[node] = 1.0f / (1.0f + expf(-o));
}

// ---------------------------------------------------------------------------
// Host-side launch
// ---------------------------------------------------------------------------
static inline size_t alignUp(size_t v, size_t a) { return (v + a - 1) & ~(a - 1); }

extern "C" void kernel_launch(void* const* d_in, const int* in_sizes, int n_in,
                              void* d_out, int out_size, void* d_ws, size_t ws_size,
                              hipStream_t stream) {
    (void)n_in; (void)out_size; (void)ws_size;
    const float*     x    = (const float*)d_in[0];
    const long long* ei   = (const long long*)d_in[1];  // int64 [2,E]
    const float*     W_in = (const float*)d_in[2];
    const float*     b_in = (const float*)d_in[3];
    const float*     msgW = (const float*)d_in[4];
    const float*     msgb = (const float*)d_in[5];
    const float*     updW = (const float*)d_in[6];
    const float*     updb = (const float*)d_in[7];
    const float*     lng  = (const float*)d_in[8];
    const float*     lnb  = (const float*)d_in[9];
    const float*     oW1  = (const float*)d_in[10];
    const float*     ob1  = (const float*)d_in[11];
    const float*     oW2  = (const float*)d_in[12];
    const float*     ob2  = (const float*)d_in[13];
    float* out = (float*)d_out;

    const int       Nn = in_sizes[0] / 39;
    const long long E  = in_sizes[1] / 2;
    const int nLayers = 2;

    // Workspace layout
    char* p = (char*)d_ws;
    float* h = (float*)p;              p += alignUp((size_t)Nn * 64 * 4, 256);
    unsigned short* hb = (unsigned short*)p; p += alignUp((size_t)Nn * 64 * 2, 256);
    float* aggr = (float*)p;           p += alignUp((size_t)Nn * 64 * 4, 256);
    float* cnt = (float*)p;            p += alignUp((size_t)Nn * 4, 256);
    unsigned short* wT = (unsigned short*)p;  // [2 layers][2 mats][64][128]

    const long long* row = ei;
    const long long* col = ei + E;

    hipMemsetAsync(cnt, 0, (size_t)Nn * 4, stream);

    int wtot = nLayers * 2 * 64 * 128;
    k_wprep<<<(wtot + 255) / 256, 256, 0, stream>>>(msgW, updW, wT, wtot);

    k_in<<<(int)(((long long)Nn * 64 + 255) / 256), 256, 0, stream>>>(
        x, W_in, b_in, h, hb, Nn);

    k_degree<<<(int)((E + 255) / 256), 256, 0, stream>>>(col, cnt, E);

    const long long edgesPerBlock = 128 * EDGE_TILES;
    for (int l = 0; l < nLayers; ++l) {
        hipMemsetAsync(aggr, 0, (size_t)Nn * 64 * 4, stream);
        k_edge<<<(int)((E + edgesPerBlock - 1) / edgesPerBlock), 256, 0, stream>>>(
            hb, row, col, wT + (size_t)(l * 2 + 0) * 64 * 128, msgb + l * 64,
            aggr, E);
        k_update<<<(Nn + 127) / 128, 256, 0, stream>>>(
            h, hb, aggr, cnt, wT + (size_t)(l * 2 + 1) * 64 * 128, updb + l * 64,
            lng + l * 64, lnb + l * 64, Nn);
    }

    k_out<<<(Nn + 255) / 256, 256, 0, stream>>>(h, oW1, ob1, oW2, ob2, out, Nn);
}
